// LSTM_38096359916293
// MI455X (gfx1250) — 16-bit and 8-bit matrix operand order repaired and checked by reading, compiled, not run on hardware
//
#include <hip/hip_runtime.h>

typedef __attribute__((ext_vector_type(16))) _Float16 v16h;
typedef __attribute__((ext_vector_type(8)))  float    v8f;
typedef __attribute__((ext_vector_type(8)))  int      v8i;

union AFrag16 { v16h v; float4 q[2]; };
union Frag8   { v8i  v; int4  q[2]; int2 d[4]; };

#define HP   256
#define GP   1024
#define HREAL 250

__device__ __host__ inline unsigned char f32_to_e4m3(float x) {
  unsigned u; __builtin_memcpy(&u, &x, 4);
  unsigned s = (u >> 24) & 0x80u;
  unsigned au = u & 0x7FFFFFFFu;
  float a; __builtin_memcpy(&a, &au, 4);
  if (a != a || a == 0.0f) return (unsigned char)s;
  if (a >= 448.0f) return (unsigned char)(s | 0x7E);
  if (a < 0.015625f) {
    int m = (int)(a * 512.0f + 0.5f);
    if (m > 7) return (unsigned char)(s | 0x08);
    return (unsigned char)(s | (unsigned)m);
  }
  int e = ((int)((au >> 23) & 0xFF)) - 127;
  float scaled = a * exp2f((float)(3 - e));
  int q = (int)(scaled + 0.5f);
  int E = e + 7;
  if (q >= 16) { q = 8; E += 1; }
  if (E >= 16 || (E == 15 && (q & 7) == 7)) return (unsigned char)(s | 0x7E);
  return (unsigned char)(s | (unsigned)(E << 3) | (unsigned)(q & 7));
}

__global__ void pack_fp8_k(const float* __restrict__ W, unsigned char* __restrict__ out) {
  int idx = blockIdx.x * blockDim.x + threadIdx.x;
  if (idx >= GP * HP) return;
  int g = idx >> 8, k = idx & 255;
  int gate = g >> 8, within = g & 255;
  float v = 0.0f;
  if (within < HREAL && k < HREAL) v = W[(gate * HREAL + within) * HREAL + k] * 16.0f;
  out[idx] = f32_to_e4m3(v);
}

__global__ void pack_f16_k(const float* __restrict__ W, _Float16* __restrict__ out) {
  int idx = blockIdx.x * blockDim.x + threadIdx.x;
  if (idx >= GP * HP) return;
  int g = idx >> 8, k = idx & 255;
  int gate = g >> 8, within = g & 255;
  float v = 0.0f;
  if (within < HREAL && k < HREAL) v = W[(gate * HREAL + within) * HREAL + k];
  out[idx] = (_Float16)v;
}

__global__ void pack_bias_k(const float* __restrict__ a, const float* __restrict__ b,
                            float* __restrict__ out) {
  int g = blockIdx.x * blockDim.x + threadIdx.x;
  if (g >= GP) return;
  int gate = g >> 8, within = g & 255;
  out[g] = (within < HREAL) ? (a[gate * HREAL + within] + b[gate * HREAL + within]) : 0.0f;
}

__global__ void pack_wih0_k(const float* __restrict__ w, float* __restrict__ out) {
  int g = blockIdx.x * blockDim.x + threadIdx.x;
  if (g >= GP) return;
  int gate = g >> 8, within = g & 255;
  out[g] = (within < HREAL) ? w[gate * HREAL + within] : 0.0f;
}

__global__ void xw0_k(const float* __restrict__ x, const float* __restrict__ w0,
                      float* __restrict__ xW, int T) {
  int idx = blockIdx.x * blockDim.x + threadIdx.x;
  if (idx >= T * GP) return;
  int t = idx >> 10, g = idx & (GP - 1);
  xW[idx] = x[t] * w0[g];
}

__global__ __launch_bounds__(256) void gemm_xw_k(const _Float16* __restrict__ hprev,
                                                 const _Float16* __restrict__ wih,
                                                 float* __restrict__ xW) {
  const int tid = threadIdx.x, wave = tid >> 5, lane = tid & 31;
  const int l15 = lane & 15, lhalf = lane >> 4;
  const int half8 = lhalf * 8, khalf16 = lhalf * 16;
  const int mt = blockIdx.x;
  const int row = mt * 16 + l15;
  const _Float16* hrow = hprev + (size_t)row * HP;
  const int nt0 = blockIdx.y * 32 + wave * 4;

  v8f acc[4];
#pragma unroll
  for (int n = 0; n < 4; ++n)
#pragma unroll
    for (int i = 0; i < 8; ++i) acc[n][i] = 0.0f;

#pragma unroll
  for (int c = 0; c < 8; ++c) {
    const int kb = c * 32;
    AFrag16 a;
    a.q[0] = *(const float4*)(hrow + kb + half8);
    a.q[1] = *(const float4*)(hrow + kb + 16 + half8);
#pragma unroll
    for (int n = 0; n < 4; ++n) {
      const _Float16* wr = wih + (size_t)((nt0 + n) * 16 + l15) * HP + kb + half8;
      AFrag16 b;
      b.q[0] = ((const float4*)wr)[0];
      b.q[1] = ((const float4*)wr)[2];
      acc[n] = __builtin_amdgcn_wmma_f32_16x16x32_f16(false, a.v, false, b.v,
                                                      (short)0, acc[n], false, false);
    }
  }
  const int r0 = mt * 16 + lhalf * 8;
#pragma unroll
  for (int n = 0; n < 4; ++n) {
    const int col = (nt0 + n) * 16 + l15;
#pragma unroll
    for (int r = 0; r < 8; ++r) xW[(size_t)(r0 + r) * GP + col] = acc[n][r];
  }
}

__global__ __launch_bounds__(1024) void lstm_scan_k(const unsigned char* __restrict__ Whh8,
                                                    const float* __restrict__ bias,
                                                    const float* __restrict__ xW,
                                                    _Float16* __restrict__ hout, int T) {
  __shared__ __align__(16) unsigned char h8sh[HP];
  __shared__ float gsh[GP];

  const int tid = threadIdx.x;
  const int wave = tid >> 5, lane = tid & 31;
  const int l15 = lane & 15, lhalf = lane >> 4;
  const int half8 = lhalf * 8, khalf16 = lhalf * 16;

  Frag8 w[2][4];
#pragma unroll
  for (int mt = 0; mt < 2; ++mt) {
    const unsigned char* wr = Whh8 + (size_t)((wave * 2 + mt) * 16 + l15) * HP;
#pragma unroll
    for (int c = 0; c < 4; ++c) {
      const int kb = c * 64;
      w[mt][c].q[0] = *(const int4*)(wr + kb + khalf16);
      w[mt][c].q[1] = *(const int4*)(wr + kb + 32 + khalf16);
    }
  }

  float c0 = 0.0f, bi = 0.f, bf = 0.f, bg = 0.f, bo = 0.f;
  if (tid < HP) {
    h8sh[tid] = 0;
    bi = bias[tid]; bf = bias[256 + tid]; bg = bias[512 + tid]; bo = bias[768 + tid];
  }
  __syncthreads();

  for (int t = 0; t < T; ++t) {
    v8f acc0, acc1;
#pragma unroll
    for (int i = 0; i < 8; ++i) { acc0[i] = 0.0f; acc1[i] = 0.0f; }

#pragma unroll
    for (int c = 0; c < 4; ++c) {
      const int kb = c * 64;
      Frag8 b;
      b.q[0] = *(const int4*)(h8sh + kb + khalf16);
      b.q[1] = *(const int4*)(h8sh + kb + 32 + khalf16);
      acc0 = __builtin_amdgcn_wmma_f32_16x16x64_fp8_fp8(w[0][c].v, b.v, (short)0, acc0, false, false);
      acc1 = __builtin_amdgcn_wmma_f32_16x16x64_fp8_fp8(w[1][c].v, b.v, (short)0, acc1, false, false);
    }
    if (l15 == 0) {
      const int g0 = (wave * 2) * 16 + lhalf * 8;
      const int g1 = (wave * 2 + 1) * 16 + lhalf * 8;
#pragma unroll
      for (int r = 0; r < 8; ++r) { gsh[g0 + r] = acc0[r]; gsh[g1 + r] = acc1[r]; }
    }
    __syncthreads();

    if (tid < HP) {
      const float inv = 0.0625f;
      const float* xwt = xW + (size_t)t * GP;
      float gi = gsh[tid]       * inv + bi + xwt[tid];
      float gf = gsh[256 + tid] * inv + bf + xwt[256 + tid];
      float gg = gsh[512 + tid] * inv + bg + xwt[512 + tid];
      float go = gsh[768 + tid] * inv + bo + xwt[768 + tid];
      float si = 1.0f / (1.0f + __expf(-gi));
      float sf = 1.0f / (1.0f + __expf(-gf));
      float tg = tanhf(gg);
      float so = 1.0f / (1.0f + __expf(-go));
      c0 = sf * c0 + si * tg;
      float h = so * tanhf(c0);
      h8sh[tid] = f32_to_e4m3(h);
      hout[(size_t)t * HP + tid] = (_Float16)h;
      if (t + 1 < T) __builtin_prefetch(xwt + GP + tid, 0, 0);
    }
    __syncthreads();
  }
}

__global__ void final_k(const _Float16* __restrict__ hlast, const float* __restrict__ Wl,
                        const float* __restrict__ bl, float* __restrict__ out) {
  __shared__ float red[256];
  int tid = threadIdx.x;
  red[tid] = (tid < HREAL) ? (float)hlast[tid] * Wl[tid] : 0.0f;
  __syncthreads();
  for (int s = 128; s > 0; s >>= 1) {
    if (tid < s) red[tid] += red[tid + s];
    __syncthreads();
  }
  if (tid == 0) out[0] = red[0] + bl[0];
}

extern "C" void kernel_launch(void* const* d_in, const int* in_sizes, int n_in,
                              void* d_out, int out_size, void* d_ws, size_t ws_size,
                              hipStream_t stream) {
  const float* x    = (const float*)d_in[0];
  const float* Wih0 = (const float*)d_in[1];
  const float* Whh0 = (const float*)d_in[2];
  const float* bih0 = (const float*)d_in[3];
  const float* bhh0 = (const float*)d_in[4];
  const float* Wih1 = (const float*)d_in[5];
  const float* Whh1 = (const float*)d_in[6];
  const float* bih1 = (const float*)d_in[7];
  const float* bhh1 = (const float*)d_in[8];
  const float* Wih2 = (const float*)d_in[9];
  const float* Whh2 = (const float*)d_in[10];
  const float* bih2 = (const float*)d_in[11];
  const float* bhh2 = (const float*)d_in[12];
  const float* Wl   = (const float*)d_in[13];
  const float* bl   = (const float*)d_in[14];
  float* out = (float*)d_out;
  const int T = in_sizes[0];

  char* ws = (char*)d_ws;
  const size_t szW8  = (size_t)GP * HP;
  const size_t szW16 = (size_t)GP * HP * 2;
  unsigned char* whh8_0 = (unsigned char*)(ws);
  unsigned char* whh8_1 = whh8_0 + szW8;
  unsigned char* whh8_2 = whh8_1 + szW8;
  _Float16* wih16_1 = (_Float16*)(whh8_2 + szW8);
  _Float16* wih16_2 = (_Float16*)((char*)wih16_1 + szW16);
  float* wih0p = (float*)((char*)wih16_2 + szW16);
  float* bias0 = wih0p + GP;
  float* bias1 = bias0 + GP;
  float* bias2 = bias1 + GP;
  _Float16* hA = (_Float16*)(bias2 + GP);
  _Float16* hB = hA + (size_t)T * HP;
  float* xW = (float*)(hB + (size_t)T * HP);

  const int NPK = GP * HP;
  pack_fp8_k<<<(NPK + 255) / 256, 256, 0, stream>>>(Whh0, whh8_0);
  pack_fp8_k<<<(NPK + 255) / 256, 256, 0, stream>>>(Whh1, whh8_1);
  pack_fp8_k<<<(NPK + 255) / 256, 256, 0, stream>>>(Whh2, whh8_2);
  pack_f16_k<<<(NPK + 255) / 256, 256, 0, stream>>>(Wih1, wih16_1);
  pack_f16_k<<<(NPK + 255) / 256, 256, 0, stream>>>(Wih2, wih16_2);
  pack_wih0_k<<<GP / 256, 256, 0, stream>>>(Wih0, wih0p);
  pack_bias_k<<<GP / 256, 256, 0, stream>>>(bih0, bhh0, bias0);
  pack_bias_k<<<GP / 256, 256, 0, stream>>>(bih1, bhh1, bias1);
  pack_bias_k<<<GP / 256, 256, 0, stream>>>(bih2, bhh2, bias2);

  xw0_k<<<(T * GP + 255) / 256, 256, 0, stream>>>(x, wih0p, xW, T);
  lstm_scan_k<<<1, 1024, 0, stream>>>(whh8_0, bias0, xW, hA, T);

  gemm_xw_k<<<dim3(T / 16, 2), 256, 0, stream>>>(hA, wih16_1, xW);
  lstm_scan_k<<<1, 1024, 0, stream>>>(whh8_1, bias1, xW, hB, T);

  gemm_xw_k<<<dim3(T / 16, 2), 256, 0, stream>>>(hB, wih16_2, xW);
  lstm_scan_k<<<1, 1024, 0, stream>>>(whh8_2, bias2, xW, hA, T);

  final_k<<<1, 256, 0, stream>>>(hA + (size_t)(T - 1) * HP, Wl, bl, out);
}
